// Decoder_43679817400487
// MI455X (gfx1250) — matrix operand order repaired and checked by reading, not compiled, not run on hardware
//
#include <hip/hip_runtime.h>
#include <math.h>

// ---------------- problem constants ----------------
#define VOCAB 32000
#define EMB   256
#define HID   256
#define BATCH 64
#define SRC   512
#define TGT   256
#define STEPS 255          // TGT-1 scan steps

typedef __attribute__((ext_vector_type(16))) __bf16 bf16x16;
typedef __attribute__((ext_vector_type(8)))  float  floatx8;

#define WMMA_BF16(a, b, c) \
  __builtin_amdgcn_wmma_f32_16x16x32_bf16(false, (a), false, (b), (short)0, (c), false, false)

// ---------------- bf16 helpers (bit-level) ----------------
__device__ __forceinline__ unsigned short f32_to_bf16_bits(float f) {
  union { float f; unsigned int u; } cv; cv.f = f;
  unsigned int u = cv.u;
  u += 0x7fffu + ((u >> 16) & 1u);     // round-to-nearest-even
  return (unsigned short)(u >> 16);
}
__device__ __forceinline__ float bf16_bits_to_f32(unsigned int s) {
  union { unsigned int u; float f; } cv; cv.u = (s & 0xffffu) << 16; return cv.f;
}
__device__ __forceinline__ __bf16 bits_to_bf16(unsigned short s) {
  union { unsigned short u; __bf16 b; } cv; cv.u = s; return cv.b;
}
__device__ __forceinline__ float sigmoidf(float x) { return 1.0f / (1.0f + __expf(-x)); }

// ---------------- WMMA fragment loaders ----------------
// A tile (16x32 bf16), row-major source: lane m = L%16; pair p:
// K = (p>=4?16:0) + (L/16)*8 + (p%4)*2 (+0,+1)
__device__ __forceinline__ bf16x16 load_a_bf16(const unsigned short* base, int lda, int m0, int k0) {
  int lane = threadIdx.x & 31;
  int m    = m0 + (lane & 15);
  int half = lane >> 4;
  const unsigned short* row = base + (size_t)m * lda + k0;
  bf16x16 a;
#pragma unroll
  for (int p = 0; p < 8; ++p) {
    int k = ((p >> 2) << 4) + (half << 3) + ((p & 3) << 1);
    unsigned int u = *(const unsigned int*)(row + k);
    a[2 * p]     = bits_to_bf16((unsigned short)(u & 0xffffu));
    a[2 * p + 1] = bits_to_bf16((unsigned short)(u >> 16));
  }
  return a;
}
// B tile (32x16 bf16) where B[k][n] = W[n][k], W row-major (ldw):
// lane n = L%16; pair p: K = (p>=4?16:0) + (L/16)*8 + (p%4)*2 (+0,+1)
__device__ __forceinline__ bf16x16 load_b_from_wT(const unsigned short* W, int ldw, int n0, int k0) {
  int lane = threadIdx.x & 31;
  int n    = n0 + (lane & 15);
  int half = lane >> 4;
  const unsigned short* row = W + (size_t)n * ldw + k0;
  bf16x16 b;
#pragma unroll
  for (int p = 0; p < 8; ++p) {
    unsigned int u = *(const unsigned int*)(row + ((p >> 2) << 4) + (half << 3) + ((p & 3) << 1));
    b[2 * p]     = bits_to_bf16((unsigned short)(u & 0xffffu));
    b[2 * p + 1] = bits_to_bf16((unsigned short)(u >> 16));
  }
  return b;
}

// ---------------- device-scope grid barrier (64 co-resident WGs) ----------------
__device__ __forceinline__ void grid_barrier(unsigned int* bar_count,
                                             volatile unsigned int* bar_gen, int nwg) {
  __threadfence();                 // drain this thread's global stores to L2
  __syncthreads();
  if (threadIdx.x == 0) {
    unsigned int g = *bar_gen;
    unsigned int arrived = atomicAdd(bar_count, 1u) + 1u;
    if (arrived == (unsigned int)nwg) {
      *bar_count = 0u;             // reset BEFORE releasing: next round is safe
      __threadfence();
      *bar_gen = g + 1u;           // release all waiters
    } else {
      while (*bar_gen == g) __builtin_amdgcn_s_sleep(2);
    }
  }
  __syncthreads();
  __threadfence();
}

// ---------------- prep kernels ----------------
__global__ void cvt_f32_bf16_kernel(const float* __restrict__ src,
                                    unsigned short* __restrict__ dst, int n) {
  int i = blockIdx.x * blockDim.x + threadIdx.x;
  int stride = gridDim.x * blockDim.x;
  for (; i < n; i += stride) dst[i] = f32_to_bf16_bits(src[i]);
}

__global__ void init_state_kernel(const float* __restrict__ dec_init,
                                  float* __restrict__ h_f32,
                                  unsigned short* __restrict__ hbf0,
                                  unsigned short* __restrict__ o_bf,
                                  unsigned int* __restrict__ bars) {
  int i = blockIdx.x * blockDim.x + threadIdx.x;
  if (i == 0) { bars[0] = 0u; bars[32] = 0u; }   // barrier count / generation
  if (i < BATCH * HID) {
    float v = dec_init[i];
    h_f32[i] = v;
    hbf0[i]  = f32_to_bf16_bits(v);
    o_bf[i]  = 0;                       // bf16 +0.0
  }
}

// proj[b,s,h] = sum_d enc[b,s,d] * W_att[h,d]   (M=32768, N=256, K=512) -> bf16
__global__ void proj_gemm_kernel(const unsigned short* __restrict__ enc_bf,
                                 const unsigned short* __restrict__ Watt_bf,
                                 unsigned short* __restrict__ proj_bf) {
  int w  = (blockIdx.x * blockDim.x + threadIdx.x) >> 5;
  int mt = w >> 4;           // 2048 m-tiles
  int nt = w & 15;           // 16 n-tiles
  floatx8 c = {};
  for (int k0 = 0; k0 < 2 * HID; k0 += 32) {
    bf16x16 a = load_a_bf16(enc_bf, 2 * HID, mt * 16, k0);
    bf16x16 b = load_b_from_wT(Watt_bf, 2 * HID, nt * 16, k0);
    c = WMMA_BF16(a, b, c);
  }
  int lane  = threadIdx.x & 31;
  int n     = nt * 16 + (lane & 15);
  int mbase = mt * 16 + ((lane >> 4) << 3);
#pragma unroll
  for (int r = 0; r < 8; ++r)
    proj_bf[(size_t)(mbase + r) * HID + n] = f32_to_bf16_bits(c[r]);
}

// G[b*255+t, n] = emb[y[b,t]] @ W_ih[:,256:].T + b_ih   (M=16320, N=768, K=256) -> f32
__global__ void embed_gemm_kernel(const float* __restrict__ emb,
                                  const int* __restrict__ y,
                                  const float* __restrict__ b_ih,
                                  const unsigned short* __restrict__ Wih_bf,
                                  float* __restrict__ G) {
  int w  = (blockIdx.x * blockDim.x + threadIdx.x) >> 5;
  int mt = w / 48;           // 1020 m-tiles
  int nt = w % 48;           // 48 n-tiles
  int lane = threadIdx.x & 31;
  int half = lane >> 4;
  int mrow = mt * 16 + (lane & 15);
  int b = mrow / STEPS, t = mrow % STEPS;
  int idx = y[b * TGT + t];
  const float* erow = emb + (size_t)idx * EMB;

  floatx8 c;
  float bias = b_ih[nt * 16 + (lane & 15)];
#pragma unroll
  for (int r = 0; r < 8; ++r) c[r] = bias;

  for (int ks = 0; ks < 8; ++ks) {
    int k0 = ks * 32;
    bf16x16 a;
#pragma unroll
    for (int p = 0; p < 8; ++p) {
      int k = k0 + ((p >> 2) << 4) + (half << 3) + ((p & 3) << 1);
      a[2 * p]     = bits_to_bf16(f32_to_bf16_bits(erow[k]));
      a[2 * p + 1] = bits_to_bf16(f32_to_bf16_bits(erow[k + 1]));
    }
    bf16x16 bb = load_b_from_wT(Wih_bf, 2 * HID, nt * 16, HID + k0); // cols 256..511
    c = WMMA_BF16(a, bb, c);
  }
  int n     = nt * 16 + (lane & 15);
  int mbase = mt * 16 + (half << 3);
#pragma unroll
  for (int r = 0; r < 8; ++r)
    G[(size_t)(mbase + r) * (3 * HID) + n] = c[r];
}

// ---------------- GRU wave body: 16 batch rows (mt) x 16 gate cols (n16) ----------
__device__ __forceinline__ void gru_wave(int mt, int n16,
                                         const unsigned short* __restrict__ o_bf,
                                         const unsigned short* __restrict__ hbf_in,
                                         unsigned short* __restrict__ hbf_out,
                                         float* __restrict__ h_f32,
                                         const unsigned short* __restrict__ Wih_bf,
                                         const unsigned short* __restrict__ Whh_bf,
                                         const float* __restrict__ G,
                                         const float* __restrict__ b_hh, int t) {
  int lane  = threadIdx.x & 31;
  int lanen = lane & 15, half = lane >> 4;
  int mbase = mt * 16 + (half << 3);

  floatx8 cgi[3], cgh[3];
#pragma unroll
  for (int g = 0; g < 3; ++g) {
    int n = g * HID + n16 * 16 + lanen;
    float bh = b_hh[n];
#pragma unroll
    for (int r = 0; r < 8; ++r) {
      int grow = (mbase + r) * STEPS + t;
      cgi[g][r] = G[(size_t)grow * (3 * HID) + n];
      cgh[g][r] = bh;
    }
  }
  for (int ks = 0; ks < 8; ++ks) {
    int k0 = ks * 32;
    bf16x16 ao = load_a_bf16(o_bf,   HID, mt * 16, k0);   // o_prev
    bf16x16 ah = load_a_bf16(hbf_in, HID, mt * 16, k0);   // h_prev
#pragma unroll
    for (int g = 0; g < 3; ++g) {
      int n0 = g * HID + n16 * 16;
      bf16x16 bi = load_b_from_wT(Wih_bf, 2 * HID, n0, k0);  // W_ih cols 0..255
      bf16x16 bh = load_b_from_wT(Whh_bf, HID,     n0, k0);
      cgi[g] = WMMA_BF16(ao, bi, cgi[g]);
      cgh[g] = WMMA_BF16(ah, bh, cgh[g]);
    }
  }
  int n = n16 * 16 + lanen;    // 0..255
#pragma unroll
  for (int r = 0; r < 8; ++r) {
    int brow = mbase + r;
    float rg = sigmoidf(cgi[0][r] + cgh[0][r]);
    float zg = sigmoidf(cgi[1][r] + cgh[1][r]);
    float ng = tanhf(cgi[2][r] + rg * cgh[2][r]);
    float ho = h_f32[brow * HID + n];
    float hn = (1.0f - zg) * ng + zg * ho;
    h_f32[brow * HID + n]           = hn;
    hbf_out[(size_t)brow * HID + n] = f32_to_bf16_bits(hn);
  }
}

// ---------------- persistent cooperative scan kernel ----------------
// Grid: 64 WGs x 512 threads. WG b owns batch row b for attention;
// WGs 0..3 additionally run the WMMA out/GRU phases for rows 16*wg..16*wg+15.
__global__ __launch_bounds__(512) void decoder_scan_kernel(
    const unsigned short* __restrict__ proj_bf,
    const unsigned short* __restrict__ enc_bf,
    const int* __restrict__ masks,
    float* __restrict__ h_f32,
    unsigned short* __restrict__ hbfA, unsigned short* __restrict__ hbfB,
    unsigned short* __restrict__ a_bf, unsigned short* __restrict__ o_bf,
    float* __restrict__ out,
    const unsigned short* __restrict__ Wcomb_bf,
    const unsigned short* __restrict__ Wih_bf,
    const unsigned short* __restrict__ Whh_bf,
    const float* __restrict__ G, const float* __restrict__ b_hh,
    unsigned int* __restrict__ bars) {
  __shared__ float sh_h[HID];
  __shared__ float salpha[SRC];
  __shared__ float red[SRC];

  int wg  = blockIdx.x;          // 0..63 == batch row
  int tid = threadIdx.x;         // 0..511
  unsigned int* bar_count = bars;
  volatile unsigned int* bar_gen = bars + 32;
  unsigned short* hbf[2] = { hbfA, hbfB };

  // ---- gru(0): o_prev = 0, h_prev = dec_init (hbf[0]) -> hbf[1], h_f32 ----
  if (wg < 4)
    gru_wave(wg, tid >> 5, o_bf, hbf[0], hbf[1], h_f32,
             Wih_bf, Whh_bf, G, b_hh, 0);
  grid_barrier(bar_count, bar_gen, BATCH);

#pragma unroll 1
  for (int t = 0; t < STEPS; ++t) {
    // ================= attention for batch row wg (uses h(t) in h_f32) =========
    {
      if (tid < HID) sh_h[tid] = h_f32[wg * HID + tid];
      __syncthreads();

      float e;
      if (masks[wg * SRC + tid] != 0) {
        e = -INFINITY;
      } else {
        const unsigned short* prow = proj_bf + ((size_t)wg * SRC + tid) * HID;
        float acc = 0.0f;
#pragma unroll 4
        for (int k = 0; k < HID; k += 2) {
          unsigned int u = *(const unsigned int*)(prow + k);
          acc += bf16_bits_to_f32(u) * sh_h[k] + bf16_bits_to_f32(u >> 16) * sh_h[k + 1];
        }
        e = acc;
      }
      red[tid] = e;
      __syncthreads();
      for (int off = 256; off > 0; off >>= 1) {
        if (tid < off) red[tid] = fmaxf(red[tid], red[tid + off]);
        __syncthreads();
      }
      float emax = red[0];
      __syncthreads();
      float ex = __expf(e - emax);
      salpha[tid] = ex;
      red[tid]    = ex;
      __syncthreads();
      for (int off = 256; off > 0; off >>= 1) {
        if (tid < off) red[tid] += red[tid + off];
        __syncthreads();
      }
      float inv = 1.0f / red[0];

      // a[d] = sum_s alpha[s]*enc[wg,s,d]; gfx1250 prefetch ahead of the stream
      float acc = 0.0f;
      const unsigned short* ebase = enc_bf + (size_t)wg * SRC * (2 * HID) + tid;
      for (int s = 0; s < SRC; s += 8) {
        if (s + 64 < SRC)
          __builtin_prefetch(ebase + (size_t)(s + 64) * (2 * HID), 0, 1);
#pragma unroll
        for (int j = 0; j < 8; ++j)
          acc += salpha[s + j] * bf16_bits_to_f32(ebase[(size_t)(s + j) * (2 * HID)]);
      }
      a_bf[(size_t)wg * (2 * HID) + tid] = f32_to_bf16_bits(acc * inv);
    }
    grid_barrier(bar_count, bar_gen, BATCH);   // a_bf complete everywhere

    // ================= out(t) then gru(t+1), rows 16*wg..16*wg+15 =============
    if (wg < 4) {
      unsigned short* h_cur  = hbf[(t + 1) & 1];   // h(t)
      unsigned short* h_next = hbf[t & 1];         // h(t+1)
      int mt = wg;
      int w  = tid >> 5;                           // 0..15
      int lane  = tid & 31;
      int lanen = lane & 15;
      int mbase = mt * 16 + ((lane >> 4) << 3);
      {
        int nt = w;
        floatx8 c = {};
        for (int k0 = 0; k0 < 3 * HID; k0 += 32) {
          bf16x16 a = (k0 < 2 * HID) ? load_a_bf16(a_bf,  2 * HID, mt * 16, k0)
                                     : load_a_bf16(h_cur, HID,     mt * 16, k0 - 2 * HID);
          bf16x16 bb = load_b_from_wT(Wcomb_bf, 3 * HID, nt * 16, k0);
          c = WMMA_BF16(a, bb, c);
        }
        int n = nt * 16 + lanen;
#pragma unroll
        for (int r = 0; r < 8; ++r) {
          int brow = mbase + r;
          float o = tanhf(c[r]);
          out[((size_t)brow * STEPS + t) * HID + n] = o;
          o_bf[brow * HID + n] = f32_to_bf16_bits(o);
        }
      }
      __syncthreads();                  // o_bf rows of this block visible
      if (t + 1 < STEPS)
        gru_wave(mt, w, o_bf, h_cur, h_next, h_f32,
                 Wih_bf, Whh_bf, G, b_hh, t + 1);
    }
    if (t + 1 < STEPS)
      grid_barrier(bar_count, bar_gen, BATCH);  // h(t+1) complete everywhere
  }
}

// ---------------- host launcher ----------------
extern "C" void kernel_launch(void* const* d_in, const int* in_sizes, int n_in,
                              void* d_out, int out_size, void* d_ws, size_t ws_size,
                              hipStream_t stream) {
  const float* enc      = (const float*)d_in[0];   // (64,512,512)
  const int*   masks    = (const int*)  d_in[1];   // (64,512)
  const float* dec_init = (const float*)d_in[2];   // (64,256)
  const int*   y        = (const int*)  d_in[3];   // (64,256)
  const float* emb      = (const float*)d_in[4];   // (32000,256)
  const float* W_att    = (const float*)d_in[5];   // (256,512)
  const float* W_ih     = (const float*)d_in[6];   // (768,512)
  const float* W_hh     = (const float*)d_in[7];   // (768,256)
  const float* b_ih     = (const float*)d_in[8];   // (768,)
  const float* b_hh     = (const float*)d_in[9];   // (768,)
  const float* W_comb   = (const float*)d_in[10];  // (256,768)
  float* out = (float*)d_out;

  // workspace layout (bytes)
  char* ws = (char*)d_ws;
  size_t off = 0;
  unsigned short* enc_bf   = (unsigned short*)(ws + off); off += (size_t)BATCH * SRC * 2 * HID * 2;  // 32 MB
  unsigned short* proj_bf  = (unsigned short*)(ws + off); off += (size_t)BATCH * SRC * HID * 2;      // 16 MB
  unsigned short* Wih_bf   = (unsigned short*)(ws + off); off += (size_t)3 * HID * 2 * HID * 2;
  unsigned short* Whh_bf   = (unsigned short*)(ws + off); off += (size_t)3 * HID * HID * 2;
  unsigned short* Wcomb_bf = (unsigned short*)(ws + off); off += (size_t)HID * 3 * HID * 2;
  unsigned short* Watt_bf  = (unsigned short*)(ws + off); off += (size_t)HID * 2 * HID * 2;
  float*          G        = (float*)(ws + off);          off += (size_t)BATCH * STEPS * 3 * HID * 4; // 50 MB
  float*          h_f32    = (float*)(ws + off);          off += (size_t)BATCH * HID * 4;
  unsigned short* hbfA     = (unsigned short*)(ws + off); off += (size_t)BATCH * HID * 2;
  unsigned short* hbfB     = (unsigned short*)(ws + off); off += (size_t)BATCH * HID * 2;
  unsigned short* a_bf     = (unsigned short*)(ws + off); off += (size_t)BATCH * 2 * HID * 2;
  unsigned short* o_bf     = (unsigned short*)(ws + off); off += (size_t)BATCH * HID * 2;
  unsigned int*   bars     = (unsigned int*)(ws + off);   off += 256;   // [0]=count, [32]=gen
  (void)ws_size; (void)n_in; (void)in_sizes; (void)out_size;

  // 1) bf16 conversions of static operands
  {
    int n = BATCH * SRC * 2 * HID;
    cvt_f32_bf16_kernel<<<dim3((n + 255) / 256), dim3(256), 0, stream>>>(enc, enc_bf, n);
    n = HID * 2 * HID;
    cvt_f32_bf16_kernel<<<dim3((n + 255) / 256), dim3(256), 0, stream>>>(W_att, Watt_bf, n);
    n = 3 * HID * 2 * HID;
    cvt_f32_bf16_kernel<<<dim3((n + 255) / 256), dim3(256), 0, stream>>>(W_ih, Wih_bf, n);
    n = 3 * HID * HID;
    cvt_f32_bf16_kernel<<<dim3((n + 255) / 256), dim3(256), 0, stream>>>(W_hh, Whh_bf, n);
    n = HID * 3 * HID;
    cvt_f32_bf16_kernel<<<dim3((n + 255) / 256), dim3(256), 0, stream>>>(W_comb, Wcomb_bf, n);
  }
  // 2) state init (h=dec_init into hbfA/h_f32, o_prev=0, barrier counters=0)
  init_state_kernel<<<dim3((BATCH * HID + 255) / 256), dim3(256), 0, stream>>>(
      dec_init, h_f32, hbfA, o_bf, bars);

  // 3) proj = enc @ W_att^T   (32768 tiles -> 4096 blocks x 8 waves)
  proj_gemm_kernel<<<dim3(4096), dim3(256), 0, stream>>>(enc_bf, Watt_bf, proj_bf);

  // 4) G = emb[y[:, :-1]] @ W_ih[:,256:]^T + b_ih  (48960 tiles -> 6120 blocks x 8 waves)
  embed_gemm_kernel<<<dim3(6120), dim3(256), 0, stream>>>(emb, y, b_ih, Wih_bf, G);

  // 5) the whole 255-step scan: one persistent cooperative kernel
  decoder_scan_kernel<<<dim3(BATCH), dim3(SRC), 0, stream>>>(
      proj_bf, enc_bf, masks, h_f32, hbfA, hbfB, a_bf, o_bf, out,
      Wcomb_bf, Wih_bf, Whh_bf, G, b_hh, bars);
}
